// ShiftConv_50474455663007
// MI455X (gfx1250) — matrix operand order repaired and checked by reading, not compiled, not run on hardware
//
#include <hip/hip_runtime.h>

typedef __bf16 bf16_t;
typedef __attribute__((ext_vector_type(16))) __bf16 v16bf;
typedef __attribute__((ext_vector_type(8)))  __bf16 v8bf;
typedef __attribute__((ext_vector_type(2)))  __bf16 v2bf;
typedef __attribute__((ext_vector_type(8)))  float   v8f;
typedef __attribute__((ext_vector_type(4)))  float   v4f;

#define TILE_M 128
#define TILE_N 128
#define TILE_K 32
#define C_IN   512
#define C_OUT  512
#define Hdim   28
#define Wdim   28
#define HW     784
#define NPOS   (64 * HW)   // 50176

// Pack two f32 -> two bf16 in one DWORD via native converts
// (lets the backend use packed cvt hardware instead of bit-twiddling).
__device__ __forceinline__ unsigned pack_bf2(float f0, float f1) {
  v2bf p;
  p.x = (__bf16)f0;
  p.y = (__bf16)f1;
  return __builtin_bit_cast(unsigned, p);
}

__global__ __launch_bounds__(256) void ShiftConv_50474455663007_kernel(
    const float* __restrict__ x, const float* __restrict__ Wm,
    const float* __restrict__ gamma, const float* __restrict__ beta,
    const float* __restrict__ rmean, const float* __restrict__ rvar,
    float* __restrict__ out)
{
  __shared__ __align__(16) bf16_t Alds[TILE_M * TILE_K];  // [m][k], 64B rows
  __shared__ __align__(16) bf16_t Blds[TILE_N * TILE_K];  // [n][k], 64B rows (transposed)
  __shared__ float s_scale[TILE_M];
  __shared__ float s_shift[TILE_M];

  const int tid   = threadIdx.x;
  const int mBase = blockIdx.x * TILE_M;
  const int pBase = blockIdx.y * TILE_N;

  // Fold BatchNorm into per-row scale/shift (done once per block).
  if (tid < TILE_M) {
    int o = mBase + tid;
    float sc = gamma[o] * rsqrtf(rvar[o] + 1e-5f);
    s_scale[tid] = sc;
    s_shift[tid] = beta[o] - rmean[o] * sc;
  }

  const int lane = tid & 31;
  const int wv   = tid >> 5;
  const int wm   = wv & 1;    // M offset = 64*wm
  const int wn   = wv >> 1;   // N offset = 32*wn
  const int sel  = (lane >> 4) & 1;  // half-wave select for fragment layout
  const int lrow = lane & 15;

  // B-staging ownership: one position, one 16-channel half per thread.
  const int jB    = tid >> 1;
  const int khalf = tid & 1;
  const int pB    = pBase + jB;
  const int bB    = pB / HW;
  const int rB    = pB - bB * HW;
  const int hB    = rB / Wdim;
  const int wB    = rB - hB * Wdim;

  // A-staging ownership: one weight row, one 16-col half per thread.
  const int rA = tid >> 1;

  v8f acc[4][2];
  #pragma unroll
  for (int mi = 0; mi < 4; ++mi)
    #pragma unroll
    for (int ni = 0; ni < 2; ++ni)
      acc[mi][ni] = (v8f){0.f, 0.f, 0.f, 0.f, 0.f, 0.f, 0.f, 0.f};

  for (int kt = 0; kt < C_IN / TILE_K; ++kt) {
    const int kBase = kt * TILE_K;
    __syncthreads();  // previous iteration's LDS reads done

    // ---- Stage A: W[mBase+rA][kBase + khalf*16 .. +15], f32 -> bf16
    {
      const float* src = Wm + (size_t)(mBase + rA) * C_IN + kBase + khalf * 16;
      __builtin_prefetch(src + TILE_K, 0, 1);  // next K tile -> global_prefetch
      unsigned pack[8];
      #pragma unroll
      for (int q = 0; q < 4; ++q) {
        v4f v = *(const v4f*)(src + q * 4);
        pack[2*q+0] = pack_bf2(v.x, v.y);
        pack[2*q+1] = pack_bf2(v.z, v.w);
      }
      unsigned* dst = (unsigned*)(Alds + rA * TILE_K + khalf * 16);
      *(uint4*)(dst + 0) = make_uint4(pack[0], pack[1], pack[2], pack[3]);
      *(uint4*)(dst + 4) = make_uint4(pack[4], pack[5], pack[6], pack[7]);
    }

    // ---- Stage B: shifted x, channels c0..c0+15 (single shift group), one position.
    {
      const int c0 = kBase + khalf * 16;
      const int g  = c0 >> 7;   // 128-channel shift group
      int hh = hB, ww = wB;
      if      (g == 0) hh = (hB + 1) % Hdim;          // roll -1 along H
      else if (g == 1) hh = (hB + Hdim - 1) % Hdim;   // roll +1 along H
      else if (g == 2) ww = (wB + 1) % Wdim;          // roll -1 along W
      else             ww = (wB + Wdim - 1) % Wdim;   // roll +1 along W
      const float* src = x + ((size_t)bB * C_IN + c0) * HW + hh * Wdim + ww;
      unsigned pack[8];
      #pragma unroll
      for (int q = 0; q < 8; ++q) {
        float f0 = src[(2*q + 0) * HW];
        float f1 = src[(2*q + 1) * HW];
        pack[q] = pack_bf2(f0, f1);
      }
      unsigned* dst = (unsigned*)(Blds + jB * TILE_K + khalf * 16);
      *(uint4*)(dst + 0) = make_uint4(pack[0], pack[1], pack[2], pack[3]);
      *(uint4*)(dst + 4) = make_uint4(pack[4], pack[5], pack[6], pack[7]);
    }

    __syncthreads();  // tiles visible

    // ---- Fragments (contiguous-k A/B layouts, k = 16*sel + s) + 8 WMMAs per wave
    v16bf bfrag[2];
    #pragma unroll
    for (int ni = 0; ni < 2; ++ni) {
      // B 32x16: lanes 0-15 hold K=0..15, lanes 16-31 hold K=16..31 (contiguous 32B)
      const bf16_t* bp = Blds + (wn * 32 + ni * 16 + lrow) * TILE_K + sel * 16;
      bfrag[ni] = *(const v16bf*)bp;
    }

    #pragma unroll
    for (int mi = 0; mi < 4; ++mi) {
      // A 16x32: lane<16 -> K 0..15; lane>=16 -> K 16..31 (matches B)
      const bf16_t* ap = Alds + (wm * 64 + mi * 16 + lrow) * TILE_K;
      union { v16bf v; v8bf h[2]; } af;
      af.h[0] = *(const v8bf*)(ap + sel * 16);
      af.h[1] = *(const v8bf*)(ap + sel * 16 + 8);
      #pragma unroll
      for (int ni = 0; ni < 2; ++ni) {
        acc[mi][ni] = __builtin_amdgcn_wmma_f32_16x16x32_bf16(
            false, af.v, false, bfrag[ni], (short)0, acc[mi][ni], false, false);
      }
    }
  }

  __syncthreads();

  // ---- Epilogue: BN scale/shift + ReLU, scatter per C/D layout.
  #pragma unroll
  for (int mi = 0; mi < 4; ++mi) {
    #pragma unroll
    for (int ni = 0; ni < 2; ++ni) {
      const int nloc = wn * 32 + ni * 16 + lrow;
      const int p = pBase + nloc;
      const int b = p / HW;
      const int r = p - b * HW;
      #pragma unroll
      for (int e = 0; e < 8; ++e) {
        const int mloc = wm * 64 + mi * 16 + e + (sel << 3);  // VGPR e -> M=e (lo) / e+8 (hi)
        const int o = mBase + mloc;
        float y = acc[mi][ni][e] * s_scale[mloc] + s_shift[mloc];
        out[((size_t)b * C_OUT + o) * HW + r] = fmaxf(y, 0.0f);
      }
    }
  }
}

extern "C" void kernel_launch(void* const* d_in, const int* in_sizes, int n_in,
                              void* d_out, int out_size, void* d_ws, size_t ws_size,
                              hipStream_t stream) {
  const float* x     = (const float*)d_in[0];
  const float* Wm    = (const float*)d_in[1];
  const float* gamma = (const float*)d_in[2];
  const float* beta  = (const float*)d_in[3];
  const float* rmean = (const float*)d_in[4];
  const float* rvar  = (const float*)d_in[5];
  float* out = (float*)d_out;

  dim3 grid(C_OUT / TILE_M, NPOS / TILE_N);  // (4, 392): M-tiles fastest for L2 reuse of x
  dim3 block(256);
  ShiftConv_50474455663007_kernel<<<grid, block, 0, stream>>>(
      x, Wm, gamma, beta, rmean, rvar, out);
}
